// AttentionBlock_26182120636494
// MI455X (gfx1250) — matrix operand order repaired and checked by reading, not compiled, not run on hardware
//
#include <hip/hip_runtime.h>

// ---------------------------------------------------------------------------
// AttentionBlock for MI455X (gfx1250): B=8, C=512, H=W=64, GROUPS=32.
// All GEMMs via v_wmma_f32_16x16x32_f16 (f16 operands, f32 accumulate).
// - Weights pre-converted once to f16 WMMA A-fragments (no cvt in hot loops).
// - Each GEMM wave computes a 16x64 strip: 1 A-frag load feeds 4 WMMAs, with
//   the 4 B-fragments explicitly multi-buffered so loads overlap the XDL pipe.
// - Attention QK pass uses two independent WMMA accumulator chains (even/odd
//   K-chunks) + software-pipelined fragment preload to avoid serial D-chains.
// - Per 16-query tile, the full 16x4096 logit row-block lives in LDS as f16
//   (128 KB of the 320 KB WGP LDS); softmax is exact and normalized in place,
//   so the P-fragment build in the PV pass is just 2x ds_load_b128 per lane.
// ---------------------------------------------------------------------------

typedef __attribute__((ext_vector_type(16))) _Float16 v16h;
typedef __attribute__((ext_vector_type(8)))  _Float16 v8h;
typedef __attribute__((ext_vector_type(8)))  float    v8f;

#define BATCH 8
#define CCH   512
#define NSP   4096                    // 64*64 spatial positions
#define GROUPS 32
#define NCHUNK_C (CCH / 32)           // 16  K-chunks over channels
#define NTILE_N  (NSP / 16)           // 256 tiles over spatial
#define NCHUNK_N (NSP / 32)           // 128 K-chunks over spatial (for V)
#define NTILE_C  (CCH / 16)           // 32  tiles over channels (for V)
#define FRAG_H   512                  // halves per 32x16 fragment (32 lanes * 16)
#define SEG ((size_t)BATCH * CCH * NSP)  // halves per f16 activation tensor
#define WSEG ((size_t)CCH * CCH)         // halves per f16 weight tensor

// --- WMMA fragment element maps (ISA 7.12.2, 16-bit A 16x32 / B 32x16) -----
__device__ __forceinline__ int bmap(int k, int j) {
    return (((k >> 3) & 1) * 16 + j) * 16 + ((k & 7) | (((k >> 4) & 1) << 3));
}
__device__ __forceinline__ int amap(int k, int m) {
    int lane = ((k >> 3) & 1) * 16 + m;
    int h    = (k & 7) | (((k >> 4) & 1) << 3);
    return lane * 16 + h;
}

// Fragment offsets (in halves) inside a tensor stored as fragments.
__device__ __forceinline__ size_t bfrag_off(int b, int c, int n) {
    size_t frag = ((size_t)b * NCHUNK_C + (c >> 5)) * NTILE_N + (n >> 4);
    return frag * FRAG_H + bmap(c & 31, n & 15);
}
__device__ __forceinline__ size_t afrag_off(int b, int c, int n) {
    size_t frag = ((size_t)b * NCHUNK_C + (c >> 5)) * NTILE_N + (n >> 4);
    return frag * FRAG_H + amap(c & 31, n & 15);
}
__device__ __forceinline__ size_t vfrag_off(int b, int c, int n) {
    size_t frag = ((size_t)b * NCHUNK_N + (n >> 5)) * NTILE_C + (c >> 4);
    return frag * FRAG_H + bmap(n & 31, c & 15);
}

__device__ __forceinline__ v8f wmma_f16(v16h a, v16h b, v8f c) {
    return __builtin_amdgcn_wmma_f32_16x16x32_f16(false, a, false, b,
                                                  (short)0, c, false, false);
}

// ---------------------------------------------------------------------------
// Kernel 0: weight prep. W (512x512 f32 [o][c]) -> f16 A-fragments laid out
// [o_tile][c_chunk][512 halves].
// ---------------------------------------------------------------------------
__global__ __launch_bounds__(256) void wprep_kernel(
    const float* __restrict__ W, _Float16* __restrict__ WF) {
    const int ot = blockIdx.x;                    // 0..31 (16-row tile)
    const int t = threadIdx.x;
    #pragma unroll 4
    for (int i = 0; i < 32; ++i) {                // 16 rows * 512 cols / 256
        int e = t + i * 256;
        int c = e & 511;
        int o = ot * 16 + (e >> 9);
        float w = W[(size_t)o * CCH + c];
        WF[((size_t)ot * 16 + (c >> 5)) * FRAG_H + amap(c & 31, o & 15)] =
            (_Float16)w;
    }
}

// ---------------------------------------------------------------------------
// Kernel 1: GroupNorm. One block per (batch, group): 16 ch x 4096 = 64K f32.
// ---------------------------------------------------------------------------
__global__ __launch_bounds__(256) void gn_kernel(
    const float* __restrict__ x, const float* __restrict__ gamma,
    const float* __restrict__ beta, _Float16* __restrict__ xn) {
    const int bg = blockIdx.x;            // 0..255
    const int b = bg >> 5, g = bg & 31;
    const int t = threadIdx.x;
    const size_t base = ((size_t)b * CCH + g * 16) * NSP;
    const float4* x4 = (const float4*)(x + base);

    float s = 0.f, ss = 0.f;
    #pragma unroll 4
    for (int i = 0; i < 64; ++i) {
        float4 v = x4[t + i * 256];
        s  += v.x + v.y + v.z + v.w;
        ss += v.x * v.x + v.y * v.y + v.z * v.z + v.w * v.w;
    }
    __shared__ float rs[256], rss[256];
    rs[t] = s; rss[t] = ss;
    __syncthreads();
    for (int off = 128; off > 0; off >>= 1) {
        if (t < off) { rs[t] += rs[t + off]; rss[t] += rss[t + off]; }
        __syncthreads();
    }
    const float mean = rs[0] * (1.f / 65536.f);
    const float var  = rss[0] * (1.f / 65536.f) - mean * mean;
    const float rinv = rsqrtf(var + 1e-5f);

    #pragma unroll 2
    for (int i = 0; i < 64; ++i) {
        int e4 = t + i * 256;
        float4 v = x4[e4];
        int e = e4 * 4;
        int c = g * 16 + (e >> 12);
        int n = e & 4095;
        float sc = gamma[c] * rinv;
        float bi = beta[c] - mean * sc;
        xn[bfrag_off(b, c, n + 0)] = (_Float16)(v.x * sc + bi);
        xn[bfrag_off(b, c, n + 1)] = (_Float16)(v.y * sc + bi);
        xn[bfrag_off(b, c, n + 2)] = (_Float16)(v.z * sc + bi);
        xn[bfrag_off(b, c, n + 3)] = (_Float16)(v.w * sc + bi);
    }
}

// ---------------------------------------------------------------------------
// Kernel 2: projection GEMM  Y[o,n] = sum_c W[o,c]*X[c,n] + bias[o].
// One wave per 16x64 strip; 4 B-fragments buffered ahead of the WMMA burst.
// mode: 0 -> B-fragments (K), 1 -> V-fragments, 2 -> A-fragments (Q)
// ---------------------------------------------------------------------------
__global__ __launch_bounds__(256) void proj_gemm(
    const _Float16* __restrict__ X, const _Float16* __restrict__ WF,
    const float* __restrict__ bias, _Float16* __restrict__ Y, int mode) {
    const int lane = threadIdx.x & 31;
    const int wid  = threadIdx.x >> 5;            // 8 waves
    const int nt0 = (blockIdx.x * 8 + wid) * 4;   // strip of 4 spatial tiles
    const int ot  = blockIdx.y;                   // output-channel tile 0..31
    const int b   = blockIdx.z;
    const int m   = lane & 15, kg = lane >> 4;

    const _Float16* wf = WF + (size_t)ot * 16 * FRAG_H + lane * 16;
    const _Float16* xb =
        X + ((size_t)b * NCHUNK_C * NTILE_N + nt0) * FRAG_H + lane * 16;

    v8f acc0 = {}, acc1 = {}, acc2 = {}, acc3 = {};
    #pragma unroll
    for (int ch = 0; ch < 16; ++ch) {
        const _Float16* xc = xb + (size_t)ch * NTILE_N * FRAG_H;
        __builtin_prefetch(xc + NTILE_N * FRAG_H, 0, 1);
        // buffer all operands for this chunk, then burst 4 WMMAs
        v16h af  = *(const v16h*)(wf + (size_t)ch * FRAG_H);
        v16h bf0 = *(const v16h*)(xc);
        v16h bf1 = *(const v16h*)(xc + 1 * FRAG_H);
        v16h bf2 = *(const v16h*)(xc + 2 * FRAG_H);
        v16h bf3 = *(const v16h*)(xc + 3 * FRAG_H);
        acc0 = wmma_f16(af, bf0, acc0);
        acc1 = wmma_f16(af, bf1, acc1);
        acc2 = wmma_f16(af, bf2, acc2);
        acc3 = wmma_f16(af, bf3, acc3);
    }
    v8f accs[4] = {acc0, acc1, acc2, acc3};
    #pragma unroll
    for (int s = 0; s < 4; ++s) {
        #pragma unroll
        for (int r = 0; r < 8; ++r) {
            int o = ot * 16 + r + 8 * kg;         // C/D: M = r + 8*(lane/16)
            int n = (nt0 + s) * 16 + m;           //      N = lane%16
            _Float16 y = (_Float16)(accs[s][r] + bias[o]);
            size_t off = (mode == 0) ? bfrag_off(b, o, n)
                       : (mode == 1) ? vfrag_off(b, o, n)
                                     : afrag_off(b, o, n);
            Y[off] = y;
        }
    }
}

// ---------------------------------------------------------------------------
// Kernel 3: attention. One block = (batch, 16-query tile). 8 waves.
// Phase 1 : S = scale * Q^T K -> 16x4096 f16 row-block in LDS. Two independent
//           accumulator chains + pipelined fragment preload.
// Phase 2 : exact softmax via LDS reduction, normalized in place.
// Phase 3 : O = P V; P fragment = 2x ds_load_b128; 4 V-frags buffered.
// ---------------------------------------------------------------------------
__global__ __launch_bounds__(256) void attn_kernel(
    const _Float16* __restrict__ Q, const _Float16* __restrict__ K,
    const _Float16* __restrict__ V, _Float16* __restrict__ AO) {
    extern __shared__ float smem[];
    _Float16* S = (_Float16*)smem;                // [16][4096] f16 logits -> P
    float* red  = smem + (16 * NSP) / 2;          // 256 f32 reduce scratch

    const int t = threadIdx.x, lane = t & 31, wid = t >> 5;
    const int qt = blockIdx.x;                    // query tile 0..255
    const int b  = blockIdx.y;
    const int m  = lane & 15, kg = lane >> 4;
    const float scale = 0.044194173824159216f;    // 512^-0.5

    // Q tile (A operand), 16 chunks held in registers for all of phase 1.
    const _Float16* qbase =
        Q + ((size_t)b * NCHUNK_C * NTILE_N + qt) * FRAG_H + lane * 16;
    v16h qf[16];
    #pragma unroll
    for (int ch = 0; ch < 16; ++ch)
        qf[ch] = *(const v16h*)(qbase + (size_t)ch * NTILE_N * FRAG_H);

    // ---- Phase 1: logits into LDS (f16) --------------------------------
    for (int jt = wid; jt < NTILE_N; jt += 8) {
        const _Float16* kbase =
            K + ((size_t)b * NCHUNK_C * NTILE_N + jt) * FRAG_H + lane * 16;
        __builtin_prefetch(kbase + 8 * (size_t)NTILE_N * FRAG_H, 0, 1);
        // two independent accumulation chains, pipelined fragment preload
        v8f acc0 = {}, acc1 = {};
        v16h bf = *(const v16h*)(kbase);
        #pragma unroll
        for (int ch = 0; ch < 16; ch += 2) {
            v16h bfn = *(const v16h*)(kbase + (size_t)(ch + 1) * NTILE_N * FRAG_H);
            acc0 = wmma_f16(qf[ch], bf, acc0);
            if (ch + 2 < 16)
                bf = *(const v16h*)(kbase + (size_t)(ch + 2) * NTILE_N * FRAG_H);
            acc1 = wmma_f16(qf[ch + 1], bfn, acc1);
        }
        #pragma unroll
        for (int r = 0; r < 8; ++r)
            S[(r + 8 * kg) * NSP + jt * 16 + m] =
                (_Float16)((acc0[r] + acc1[r]) * scale);
    }
    __syncthreads();

    // ---- Phase 2: softmax (16 threads per row), normalize in place -----
    {
        const int row = t >> 4, sub = t & 15;
        _Float16* sr = S + row * NSP;
        float mx = -3.0e38f;
        for (int j = sub; j < NSP; j += 16) mx = fmaxf(mx, (float)sr[j]);
        red[t] = mx;
        __syncthreads();
        for (int off = 8; off > 0; off >>= 1) {
            if (sub < off) red[t] = fmaxf(red[t], red[t + off]);
            __syncthreads();
        }
        mx = red[row * 16];
        float sum = 0.f;
        for (int j = sub; j < NSP; j += 16) sum += __expf((float)sr[j] - mx);
        __syncthreads();                           // red[] reuse
        red[t] = sum;
        __syncthreads();
        for (int off = 8; off > 0; off >>= 1) {
            if (sub < off) red[t] += red[t + off];
            __syncthreads();
        }
        const float inv = 1.f / red[row * 16];
        // In-place: element (row, j) read+written only by thread (row, j%16).
        for (int j = sub; j < NSP; j += 16)
            sr[j] = (_Float16)(__expf((float)sr[j] - mx) * inv);
    }
    __syncthreads();

    // ---- Phase 3: O = P V; wave wid owns channels [wid*64, wid*64+64) ---
    v8f oacc0 = {}, oacc1 = {}, oacc2 = {}, oacc3 = {};
    for (int ch = 0; ch < NCHUNK_N; ++ch) {
        // A fragment of P straight from LDS: two contiguous 16B runs (amap).
        const _Float16* sr = S + m * NSP + ch * 32 + 8 * kg;
        v8h a0 = *(const v8h*)sr;                  // K = 8*kg .. 8*kg+7
        v8h a1 = *(const v8h*)(sr + 16);           // K = 16+8*kg .. 16+8*kg+7
        v16h af;
        #pragma unroll
        for (int h = 0; h < 8; ++h) { af[h] = a0[h]; af[h + 8] = a1[h]; }

        const _Float16* vbase = V +
            ((size_t)b * NCHUNK_N + ch) * NTILE_C * FRAG_H +
            (size_t)(wid * 4) * FRAG_H + lane * 16;
        __builtin_prefetch(vbase + (size_t)NTILE_C * FRAG_H, 0, 1);
        v16h bf0 = *(const v16h*)(vbase);
        v16h bf1 = *(const v16h*)(vbase + 1 * FRAG_H);
        v16h bf2 = *(const v16h*)(vbase + 2 * FRAG_H);
        v16h bf3 = *(const v16h*)(vbase + 3 * FRAG_H);
        oacc0 = wmma_f16(af, bf0, oacc0);
        oacc1 = wmma_f16(af, bf1, oacc1);
        oacc2 = wmma_f16(af, bf2, oacc2);
        oacc3 = wmma_f16(af, bf3, oacc3);
    }
    // Store attn_out as B-fragments for the output projection.
    v8f oaccs[4] = {oacc0, oacc1, oacc2, oacc3};
    #pragma unroll
    for (int ct = 0; ct < 4; ++ct) {
        #pragma unroll
        for (int r = 0; r < 8; ++r) {
            int q = qt * 16 + r + 8 * kg;          // M dim = query
            int c = wid * 64 + ct * 16 + m;        // N dim = channel
            AO[bfrag_off(b, c, q)] = (_Float16)oaccs[ct][r];
        }
    }
}

// ---------------------------------------------------------------------------
// Kernel 4: output projection + bias + residual, f32 result. 16x64 strips.
// ---------------------------------------------------------------------------
__global__ __launch_bounds__(256) void out_proj(
    const _Float16* __restrict__ AO, const _Float16* __restrict__ WF,
    const float* __restrict__ bo, const float* __restrict__ x,
    float* __restrict__ out) {
    const int lane = threadIdx.x & 31;
    const int wid  = threadIdx.x >> 5;
    const int nt0 = (blockIdx.x * 8 + wid) * 4;
    const int ot  = blockIdx.y;
    const int b   = blockIdx.z;
    const int m   = lane & 15, kg = lane >> 4;

    const _Float16* wf = WF + (size_t)ot * 16 * FRAG_H + lane * 16;
    const _Float16* xb =
        AO + ((size_t)b * NCHUNK_C * NTILE_N + nt0) * FRAG_H + lane * 16;

    v8f acc0 = {}, acc1 = {}, acc2 = {}, acc3 = {};
    #pragma unroll
    for (int ch = 0; ch < 16; ++ch) {
        const _Float16* xc = xb + (size_t)ch * NTILE_N * FRAG_H;
        v16h af  = *(const v16h*)(wf + (size_t)ch * FRAG_H);
        v16h bf0 = *(const v16h*)(xc);
        v16h bf1 = *(const v16h*)(xc + 1 * FRAG_H);
        v16h bf2 = *(const v16h*)(xc + 2 * FRAG_H);
        v16h bf3 = *(const v16h*)(xc + 3 * FRAG_H);
        acc0 = wmma_f16(af, bf0, acc0);
        acc1 = wmma_f16(af, bf1, acc1);
        acc2 = wmma_f16(af, bf2, acc2);
        acc3 = wmma_f16(af, bf3, acc3);
    }
    v8f accs[4] = {acc0, acc1, acc2, acc3};
    #pragma unroll
    for (int s = 0; s < 4; ++s) {
        #pragma unroll
        for (int r = 0; r < 8; ++r) {
            int o = ot * 16 + r + 8 * kg;
            int n = (nt0 + s) * 16 + m;
            size_t idx = ((size_t)b * CCH + o) * NSP + n;
            out[idx] = accs[s][r] + bo[o] + x[idx];   // +residual
        }
    }
}

// ---------------------------------------------------------------------------
extern "C" void kernel_launch(void* const* d_in, const int* in_sizes, int n_in,
                              void* d_out, int out_size, void* d_ws, size_t ws_size,
                              hipStream_t stream) {
    const float* x   = (const float*)d_in[0];
    const float* gs  = (const float*)d_in[1];
    const float* gb  = (const float*)d_in[2];
    const float* wq  = (const float*)d_in[3];
    const float* bq  = (const float*)d_in[4];
    const float* wk  = (const float*)d_in[5];
    const float* bk  = (const float*)d_in[6];
    const float* wv  = (const float*)d_in[7];
    const float* bv  = (const float*)d_in[8];
    const float* wo  = (const float*)d_in[9];
    const float* bo  = (const float*)d_in[10];
    float* out = (float*)d_out;

    // Workspace carve: 5 activation tensors (32 MB each) + 4 weight tensors
    // (0.5 MB each) in f16 fragment order.  Total ~162 MB.
    _Float16* ws  = (_Float16*)d_ws;
    _Float16* xn  = ws;
    _Float16* q   = ws + 1 * SEG;
    _Float16* k   = ws + 2 * SEG;
    _Float16* v   = ws + 3 * SEG;
    _Float16* ao  = ws + 4 * SEG;
    _Float16* wqf = ws + 5 * SEG;
    _Float16* wkf = wqf + WSEG;
    _Float16* wvf = wkf + WSEG;
    _Float16* wof = wvf + WSEG;

    wprep_kernel<<<NTILE_C, 256, 0, stream>>>(wq, wqf);
    wprep_kernel<<<NTILE_C, 256, 0, stream>>>(wk, wkf);
    wprep_kernel<<<NTILE_C, 256, 0, stream>>>(wv, wvf);
    wprep_kernel<<<NTILE_C, 256, 0, stream>>>(wo, wof);

    gn_kernel<<<BATCH * GROUPS, 256, 0, stream>>>(x, gs, gb, xn);

    dim3 gg(NTILE_N / 4 / 8, NTILE_C, BATCH);      // (8, 32, 8)
    proj_gemm<<<gg, 256, 0, stream>>>(xn, wqf, bq, q, 2);   // Q -> A-frags
    proj_gemm<<<gg, 256, 0, stream>>>(xn, wkf, bk, k, 0);   // K -> B-frags
    proj_gemm<<<gg, 256, 0, stream>>>(xn, wvf, bv, v, 1);   // V -> V-frags

    // 16*4096 f16 logits (128 KB) + 1 KB reduce scratch < 320 KB WGP LDS.
    size_t smem = (size_t)(16 * NSP) * sizeof(_Float16) + 256 * sizeof(float);
    attn_kernel<<<dim3(NTILE_N, BATCH), 256, smem, stream>>>(q, k, v, ao);

    out_proj<<<gg, 256, 0, stream>>>(ao, wof, bo, x, out);
}
